// DenseJungleSubnet_2628519985538
// MI455X (gfx1250) — matrix operand order repaired and checked by reading, not compiled, not run on hardware
//
#include <hip/hip_runtime.h>

// ---------------------------------------------------------------------------
// DenseJungleSubnet on MI455X (gfx1250, wave32, WMMA, async-LDS)
//   state   : 8192 x 2112 fp32   (workspace, accumulation precision)
//   mirror  : 8192 x 2112 bf16   (workspace, WMMA A-operand feed)
//   Layer i : GEMM (M=8192, K=1024+i, N=1088-i) bf16 WMMA / fp32 accum,
//             bias + ReLU, += into state tail, refresh bf16 mirror.
// ---------------------------------------------------------------------------

typedef __attribute__((ext_vector_type(16))) __bf16 v16bf;
typedef __attribute__((ext_vector_type(8)))  float  v8f;
typedef int v4i __attribute__((vector_size(16)));   // 16-byte chunk for async-LDS builtin

#define AS1 __attribute__((address_space(1)))
#define AS3 __attribute__((address_space(3)))

#if defined(__AMDGCN__) && __has_builtin(__builtin_amdgcn_global_load_async_to_lds_b128)
#define USE_ASYNC_LDS 1
#else
#define USE_ASYNC_LDS 0
#endif

#define BATCH   8192
#define TOTAL   2112      // 1024 + 64 + 1024
#define N_IN    1024
#define N_HID   64
#define N_OUT   1024

#define BM      128
#define BN      128
#define BK      32
#define PAD_K   40        // LDS row stride in bf16 elems (80 B: 16B-aligned, bank-skewed)

__device__ __forceinline__ unsigned rne_bf16_bits(float f) {
  union { float f; unsigned u; } v; v.f = f;
  return v.u + 0x7FFFu + ((v.u >> 16) & 1u);   // bf16 in bits [31:16], RNE
}

__device__ __forceinline__ unsigned short f32_to_bf16(float f) {
  return (unsigned short)(rne_bf16_bits(f) >> 16);
}

// Pack two fp32 -> bf16x2 (RNE). Prefer v_cvt_pk_bf16_f32 if the toolchain has
// it; else round both and merge the high halves with a single v_perm_b32.
__device__ __forceinline__ unsigned pack_bf16x2(float a, float b) {
#if defined(__AMDGCN__) && __has_builtin(__builtin_amdgcn_cvt_pk_bf16_f32)
  auto t = __builtin_amdgcn_cvt_pk_bf16_f32(a, b);
  unsigned u;
  __builtin_memcpy(&u, &t, 4);
  return u;
#elif defined(__AMDGCN__) && __has_builtin(__builtin_amdgcn_perm)
  // D.byte0..3 = {S0.b2, S0.b3, S1.b2, S1.b3} -> {a_hi16 low, b_hi16 high}
  return __builtin_amdgcn_perm(rne_bf16_bits(a), rne_bf16_bits(b), 0x03020706u);
#else
  return (unsigned)f32_to_bf16(a) | ((unsigned)f32_to_bf16(b) << 16);
#endif
}

__device__ __forceinline__ void wait_async0() {
#if USE_ASYNC_LDS
#if __has_builtin(__builtin_amdgcn_s_wait_asynccnt)
  __builtin_amdgcn_s_wait_asynccnt(0);
#else
  asm volatile("s_wait_asynccnt 0x0" ::: "memory");
#endif
#endif
}

// ---- A tile staging: bf16 mirror -> LDS (128 rows x 32 K, raw bf16 bytes) ----
// Thread t copies 32 B: row = t>>1, elem offset (t&1)*16.
__device__ __forceinline__ void stage_A(const unsigned short* __restrict__ mir,
                                        int mBase, int k0, int K,
                                        unsigned short* lds, int tid) {
  const int row = tid >> 1;
  const int c0  = (tid & 1) * 16;
  const unsigned short* src = mir + (size_t)(mBase + row) * TOTAL + k0 + c0;
  unsigned short* dst = &lds[row * PAD_K + c0];
  if (k0 + BK <= K) {
#if USE_ASYNC_LDS
    __builtin_amdgcn_global_load_async_to_lds_b128(
        (AS1 v4i*)src,       (AS3 v4i*)dst,       0, 0);
    __builtin_amdgcn_global_load_async_to_lds_b128(
        (AS1 v4i*)(src + 8), (AS3 v4i*)(dst + 8), 0, 0);
#else
    *reinterpret_cast<uint4*>(dst)     = *reinterpret_cast<const uint4*>(src);
    *reinterpret_cast<uint4*>(dst + 8) = *reinterpret_cast<const uint4*>(src + 8);
#endif
  } else {
    #pragma unroll
    for (int j = 0; j < 16; ++j)
      dst[j] = (k0 + c0 + j < K) ? src[j] : (unsigned short)0;
  }
}

// ---- B tile: global fp32 -> regs (prefetch), regs -> bf16 LDS (commit) ----
struct BRegs { float4 v[4]; };

__device__ __forceinline__ void load_B_regs(const float* __restrict__ W, int ldg,
                                            int nBase, int Nout, int k0, int K,
                                            int vecOK, BRegs& t, int tid) {
  const int row  = tid >> 1;
  const int col0 = (tid & 1) * 16;
  const int grow = nBase + row;
  #pragma unroll
  for (int q = 0; q < 4; ++q) {
    float4 f = {0.f, 0.f, 0.f, 0.f};
    const int k = k0 + col0 + q * 4;
    if (grow < Nout) {
      const float* p = W + (size_t)grow * ldg + k;
      if (vecOK && (k0 + BK <= K)) {
        f = *reinterpret_cast<const float4*>(p);
      } else {
        if (k + 0 < K) f.x = p[0];
        if (k + 1 < K) f.y = p[1];
        if (k + 2 < K) f.z = p[2];
        if (k + 3 < K) f.w = p[3];
      }
    }
    t.v[q] = f;
  }
}

__device__ __forceinline__ void store_B_lds(const BRegs& t, unsigned short* lds, int tid) {
  const int row  = tid >> 1;
  const int col0 = (tid & 1) * 16;
  #pragma unroll
  for (int q = 0; q < 4; ++q) {
    uint2 p;
    p.x = pack_bf16x2(t.v[q].x, t.v[q].y);
    p.y = pack_bf16x2(t.v[q].z, t.v[q].w);
    *reinterpret_cast<uint2*>(&lds[row * PAD_K + col0 + q * 4]) = p;
  }
}

union FragU { v16bf v; uint4 q[2]; };

// One layer: state[:, colStart+n] += relu(state[:, :K] @ W[n,:K]^T + bias[n])
__global__ __launch_bounds__(256)
void jungle_layer_kernel(float* __restrict__ state,
                         unsigned short* __restrict__ mirror,
                         const float* __restrict__ W,
                         const float* __restrict__ bias,
                         int K, int Nout, int colStart, int vecOKB) {
  __shared__ unsigned short As[2][BM * PAD_K];
  __shared__ unsigned short Bs[2][BN * PAD_K];

  const int tid   = threadIdx.x;
  const int wave  = tid >> 5;
  const int lane  = tid & 31;
  const int lrow  = lane & 15;
  const int lhalf = lane >> 4;
  const int wm    = wave & 3;          // 4 wave positions along M (2 sub-tiles each)
  const int wn    = wave >> 2;         // 2 wave positions along N (4 sub-tiles each)

  const int mBase = blockIdx.y * BM;
  const int nBase = blockIdx.x * BN;

  const v8f zero = {0.f, 0.f, 0.f, 0.f, 0.f, 0.f, 0.f, 0.f};
  v8f acc[2][4];
  #pragma unroll
  for (int mi = 0; mi < 2; ++mi)
    #pragma unroll
    for (int nt = 0; nt < 4; ++nt) acc[mi][nt] = zero;

  const int nsteps = (K + BK - 1) / BK;

  // Prologue: stage step 0 into buffer 0.
  BRegs breg;
  stage_A(mirror, mBase, 0, K, As[0], tid);
  load_B_regs(W, K, nBase, Nout, 0, K, vecOKB, breg, tid);
  store_B_lds(breg, Bs[0], tid);
  wait_async0();
  __syncthreads();

  for (int s = 0; s < nsteps; ++s) {
    const int cur = s & 1;
    const bool more = (s + 1 < nsteps);

    // Prefetch step s+1 into the other buffer (overlaps with WMMA below).
    if (more) {
      stage_A(mirror, mBase, (s + 1) * BK, K, As[cur ^ 1], tid);          // async
      load_B_regs(W, K, nBase, Nout, (s + 1) * BK, K, vecOKB, breg, tid); // to regs
    }

    // Compute 2x4 sub-tiles from buffer `cur`.
    FragU a[2];
    #pragma unroll
    for (int mi = 0; mi < 2; ++mi) {
      const int arow = (wm * 2 + mi) * 16 + lrow;
      const uint4* Ap =
          reinterpret_cast<const uint4*>(&As[cur][arow * PAD_K + lhalf * 8]);
      a[mi].q[0] = Ap[0];   // K = kb .. kb+7
      a[mi].q[1] = Ap[2];   // K = kb+16 .. kb+23
    }
    #pragma unroll
    for (int nt = 0; nt < 4; ++nt) {
      const int brow = (wn * 4 + nt) * 16 + lrow;
      FragU b;
      const uint4* Bp =
          reinterpret_cast<const uint4*>(&Bs[cur][brow * PAD_K + lhalf * 8]);
      b.q[0] = Bp[0];
      b.q[1] = Bp[2];
      acc[0][nt] = __builtin_amdgcn_wmma_f32_16x16x32_bf16(
          false, a[0].v, false, b.v, (short)0, acc[0][nt], false, false);
      acc[1][nt] = __builtin_amdgcn_wmma_f32_16x16x32_bf16(
          false, a[1].v, false, b.v, (short)0, acc[1][nt], false, false);
    }

    if (more) {
      store_B_lds(breg, Bs[cur ^ 1], tid);  // other buffer: safe during compute
      wait_async0();                        // A async copies for s+1 landed
      __syncthreads();                      // everyone done with `cur`, s+1 ready
    }
  }

  // Epilogue: C/D layout -> lane gives N, VGPR r gives M (lane[4]*8 split).
  #pragma unroll
  for (int mi = 0; mi < 2; ++mi) {
    const int row0 = mBase + (wm * 2 + mi) * 16 + lhalf * 8;
    float*          prow = state  + (size_t)row0 * TOTAL + colStart;
    unsigned short* qrow = mirror + (size_t)row0 * TOTAL + colStart;
    #pragma unroll
    for (int nt = 0; nt < 4; ++nt) {
      const int n = nBase + (wn * 4 + nt) * 16 + lrow;
      if (n < Nout) {
        const float bv = bias[n];
        #pragma unroll
        for (int r = 0; r < 8; ++r) {
          float v = acc[mi][nt][r] + bv;
          v = v > 0.f ? v : 0.f;
          const float sum = prow[(size_t)(r * TOTAL) + n] + v;
          prow[(size_t)(r * TOTAL) + n] = sum;              // fp32 state
          qrow[(size_t)(r * TOTAL) + n] = f32_to_bf16(sum); // bf16 mirror
        }
      }
    }
  }
}

__global__ void jungle_init_kernel(const float* __restrict__ x,
                                   float* __restrict__ state,
                                   unsigned short* __restrict__ mirror) {
  size_t idx = (size_t)blockIdx.x * blockDim.x + threadIdx.x;
  if (idx < (size_t)BATCH * TOTAL) {
    const int row = (int)(idx / TOTAL);
    const int col = (int)(idx % TOTAL);
    const float v = (col < N_IN) ? x[(size_t)row * N_IN + col] : 0.f;
    state[idx]  = v;
    mirror[idx] = f32_to_bf16(v);
  }
}

__global__ void jungle_out_kernel(const float* __restrict__ state,
                                  float* __restrict__ out) {
  size_t idx = (size_t)blockIdx.x * blockDim.x + threadIdx.x;
  if (idx < (size_t)BATCH * N_OUT) {
    const int row = (int)(idx / N_OUT);
    const int col = (int)(idx % N_OUT);
    out[idx] = state[(size_t)row * TOTAL + (TOTAL - N_OUT) + col];
  }
}

extern "C" void kernel_launch(void* const* d_in, const int* in_sizes, int n_in,
                              void* d_out, int out_size, void* d_ws, size_t ws_size,
                              hipStream_t stream) {
  (void)in_sizes; (void)n_in; (void)out_size; (void)ws_size;
  const float* x = (const float*)d_in[0];
  float* state = (float*)d_ws;                                   // 69.2 MB fp32
  unsigned short* mirror =
      (unsigned short*)((char*)d_ws + (size_t)BATCH * TOTAL * sizeof(float)); // 34.6 MB bf16

  {
    const size_t total = (size_t)BATCH * TOTAL;
    jungle_init_kernel<<<dim3((unsigned)((total + 255) / 256)), 256, 0, stream>>>(
        x, state, mirror);
  }

  for (int i = 0; i < N_HID; ++i) {
    const float* W  = (const float*)d_in[1 + i];
    const float* bs = (const float*)d_in[1 + N_HID + i];
    const int K        = N_IN + i;              // fan_in
    const int Nout     = N_OUT + N_HID - i;     // fan_out
    const int colStart = N_IN + i;
    dim3 grid((Nout + BN - 1) / BN, BATCH / BM);
    jungle_layer_kernel<<<grid, 256, 0, stream>>>(state, mirror, W, bs, K, Nout,
                                                  colStart, (K % 4 == 0) ? 1 : 0);
  }

  {
    const size_t total = (size_t)BATCH * N_OUT;
    jungle_out_kernel<<<dim3((unsigned)((total + 255) / 256)), 256, 0, stream>>>(
        state, (float*)d_out);
  }
}
